// ImprovedGraphSAGE_3762391351471
// MI455X (gfx1250) — matrix operand order repaired and checked by reading, not compiled, not run on hardware
//
#include <hip/hip_runtime.h>
#include <hip/hip_bf16.h>
#include <math.h>

// ---------------- problem constants (from reference) ----------------
#define N_NODES 20000
#define N_EDGES 320000
#define IN_DIM  256
#define HID     128
#define OUT_DIM 64
#define EPS     1e-5f

typedef __attribute__((ext_vector_type(16))) __bf16 v16bf;
typedef __attribute__((ext_vector_type(8)))  __bf16 v8bf;
typedef __attribute__((ext_vector_type(8)))  float  v8f;

// ---------------- utility kernels ----------------
__global__ void k_cvt_bf16(const float* __restrict__ in, __bf16* __restrict__ out, int n) {
  int i = blockIdx.x * blockDim.x + threadIdx.x;
  if (i < n) out[i] = (__bf16)in[i];
}

__global__ void k_fill_f32(float* __restrict__ p, float v, int n) {
  int i = blockIdx.x * blockDim.x + threadIdx.x;
  if (i < n) p[i] = v;
}

__global__ void k_fill_u32(unsigned* __restrict__ p, unsigned v, int n) {
  int i = blockIdx.x * blockDim.x + threadIdx.x;
  if (i < n) p[i] = v;
}

// ---------------- WMMA bf16 GEMM: C[M,N] = A[M,K] @ B[K,N] (+bias) ----------------
// One wave = one 16x16 output tile. Block = 128 threads = 4 waves covering 4 N-tiles.
// A-frag (16-bit 16x32 layout): lane L holds row M=L%16; halves 0..7 = K(sel8..sel8+7),
// halves 8..15 = K(16+sel8..), sel8 = 8*(L/16).
// B-frag (32x16): lane L holds col N=ntile+L%16; halves 0..7 = K(sel8..sel8+7), halves 8..15 = K(16+sel8..).
// C/D: VGPR r -> row r + 8*(L/16), col L%16.
__global__ __launch_bounds__(128)
void k_gemm_bf16(const __bf16* __restrict__ A, const __bf16* __restrict__ B,
                 const float* __restrict__ bias, float* __restrict__ C,
                 int M, int N, int K)
{
  const int lane  = threadIdx.x & 31;
  const int wave  = threadIdx.x >> 5;
  const int mtile = blockIdx.x * 16;
  const int ntile = (blockIdx.y * 4 + wave) * 16;
  if (mtile >= M || ntile >= N) return;   // wave-uniform: EXEC stays all-ones

  const int mrowA = mtile + (lane & 15);
  const int sel8  = (lane >> 4) << 3;          // 0 or 8
  const __bf16* __restrict__ arow = A + (size_t)mrowA * K;

  v8f acc = {};
  for (int kc = 0; kc < K; kc += 32) {
    // prefetch next K-chunk of A (global_prefetch_b8)
    if (kc + 32 < K) __builtin_prefetch((const void*)(arow + kc + 32), 0, 3);

    v8bf alo = *(const v8bf*)(arow + kc + sel8);
    v8bf ahi = *(const v8bf*)(arow + kc + 16 + sel8);
    v16bf a;
#pragma unroll
    for (int i = 0; i < 8; ++i) { a[i] = alo[i]; a[i + 8] = ahi[i]; }

    v16bf b;
#pragma unroll
    for (int i = 0; i < 8; ++i) {
      b[i]     = B[(size_t)(kc + sel8 + i) * N + ntile + (lane & 15)];
      b[i + 8] = B[(size_t)(kc + 16 + sel8 + i) * N + ntile + (lane & 15)];
    }

    acc = __builtin_amdgcn_wmma_f32_16x16x32_bf16(
        /*neg_a=*/false, a, /*neg_b=*/false, b,
        /*c_mod=*/(short)0, acc, /*reuse_a=*/false, /*reuse_b=*/false);
  }

  const int ncol  = ntile + (lane & 15);
  const int mbase = mtile + ((lane >> 4) << 3);
  const float bv  = bias ? bias[ncol] : 0.0f;
#pragma unroll
  for (int r = 0; r < 8; ++r)
    C[(size_t)(mbase + r) * N + ncol] = acc[r] + bv;
}

// ---------------- attention logits: al_s[n,h] = <xh[n,h,:], a_src[h,:]> ----------------
__global__ void k_attn_logits(const float* __restrict__ xh, const float* __restrict__ a_src,
                              const float* __restrict__ a_dst, float* __restrict__ al_s,
                              float* __restrict__ al_d, int Nn, int H, int C)
{
  int i = blockIdx.x * blockDim.x + threadIdx.x;   // over Nn*H
  if (i >= Nn * H) return;
  int h = i % H, n = i / H;
  const float* __restrict__ xp = xh + ((size_t)n * H + h) * C;
  const float* __restrict__ as = a_src + (size_t)h * C;
  const float* __restrict__ ad = a_dst + (size_t)h * C;
  float s = 0.f, d = 0.f;
  for (int c = 0; c < C; ++c) { float v = xp[c]; s += v * as[c]; d += v * ad[c]; }
  al_s[i] = s; al_d[i] = d;
}

// order-preserving float<->u32 key for atomic max over signed floats
__device__ __forceinline__ unsigned f2ord(float f) {
  unsigned u = __float_as_uint(f);
  return (u & 0x80000000u) ? ~u : (u | 0x80000000u);
}
__device__ __forceinline__ float ord2f(unsigned u) {
  return (u & 0x80000000u) ? __uint_as_float(u & 0x7fffffffu) : __uint_as_float(~u);
}
__device__ __forceinline__ float lrelu(float x) { return x > 0.f ? x : 0.2f * x; }

// ---------------- edge pass 1: segment max over incoming edges ----------------
__global__ void k_edge_max(const int* __restrict__ esrc, const int* __restrict__ edst,
                           int E, int Etot, const float* __restrict__ al_s,
                           const float* __restrict__ al_d, unsigned* __restrict__ mkey, int H)
{
  int i = blockIdx.x * blockDim.x + threadIdx.x;   // over Etot*H
  if (i >= Etot * H) return;
  int h = i % H, e = i / H;
  int s = (e < E) ? esrc[e] : (e - E);
  int d = (e < E) ? edst[e] : (e - E);
  float ev = lrelu(al_s[s * H + h] + al_d[d * H + h]);
  atomicMax(&mkey[d * H + h], f2ord(ev));
}

// ---------------- edge pass 2: exp(e - m[dst]) and segment sum ----------------
__global__ void k_edge_expsum(const int* __restrict__ esrc, const int* __restrict__ edst,
                              int E, int Etot, const float* __restrict__ al_s,
                              const float* __restrict__ al_d, const unsigned* __restrict__ mkey,
                              float* __restrict__ exbuf, float* __restrict__ denom, int H)
{
  int i = blockIdx.x * blockDim.x + threadIdx.x;   // over Etot*H
  if (i >= Etot * H) return;
  int h = i % H, e = i / H;
  int s = (e < E) ? esrc[e] : (e - E);
  int d = (e < E) ? edst[e] : (e - E);
  float ev = lrelu(al_s[s * H + h] + al_d[d * H + h]);
  float ex = __expf(ev - ord2f(mkey[d * H + h]));
  exbuf[i] = ex;
  atomicAdd(&denom[d * H + h], ex);
}

// ---------------- edge pass 3: agg[dst,h,:] += xh[src,h,:] * alpha ----------------
__global__ void k_edge_agg(const int* __restrict__ esrc, const int* __restrict__ edst,
                           int E, int Etot, const float* __restrict__ exbuf,
                           const float* __restrict__ denom, const float* __restrict__ xh,
                           float* __restrict__ agg, int H, int C)
{
  long long i = (long long)blockIdx.x * blockDim.x + threadIdx.x;
  long long tot = (long long)Etot * H * C;
  if (i >= tot) return;
  int c = (int)(i % C);
  int h = (int)((i / C) % H);
  int e = (int)(i / ((long long)C * H));
  int s = (e < E) ? esrc[e] : (e - E);
  int d = (e < E) ? edst[e] : (e - E);
  float alpha = exbuf[(size_t)e * H + h] / denom[d * H + h];
  atomicAdd(&agg[((size_t)d * H + h) * C + c],
            xh[((size_t)s * H + h) * C + c] * alpha);
}

// ---------------- fused epilogue: mean-heads + bias, LayerNorm, +identity, GELU, ->bf16 ----
__global__ __launch_bounds__(128)
void k_post(const float* __restrict__ agg, const float* __restrict__ bias,
            const float* __restrict__ g, const float* __restrict__ beta,
            const float* __restrict__ identity, __bf16* __restrict__ h_bf,
            int H, int C)
{
  const int n = blockIdx.x;
  const int c = threadIdx.x;            // C == 128 == blockDim.x
  float v = 0.f;
  for (int h = 0; h < H; ++h) v += agg[((size_t)n * H + h) * C + c];
  v = v / (float)H + bias[c];

  __shared__ float red[128];
  red[c] = v; __syncthreads();
  for (int s = 64; s > 0; s >>= 1) { if (c < s) red[c] += red[c + s]; __syncthreads(); }
  float mu = red[0] / (float)C; __syncthreads();

  float dv = v - mu;
  red[c] = dv * dv; __syncthreads();
  for (int s = 64; s > 0; s >>= 1) { if (c < s) red[c] += red[c + s]; __syncthreads(); }
  float var = red[0] / (float)C;

  float y  = dv * rsqrtf(var + EPS) * g[c] + beta[c] + identity[(size_t)n * C + c];
  float ge = 0.5f * y * (1.0f + erff(y * 0.70710678118654752f));   // exact GELU
  h_bf[(size_t)n * C + c] = (__bf16)ge;
}

// ---------------- final output: H=1 head, out = agg + b2 ----------------
__global__ void k_final(const float* __restrict__ agg, const float* __restrict__ b,
                        float* __restrict__ out, int Nn, int C)
{
  int i = blockIdx.x * blockDim.x + threadIdx.x;
  if (i >= Nn * C) return;
  out[i] = agg[i] + b[i % C];
}

// =================================================================================
extern "C" void kernel_launch(void* const* d_in, const int* in_sizes, int n_in,
                              void* d_out, int out_size, void* d_ws, size_t ws_size,
                              hipStream_t stream)
{
  (void)in_sizes; (void)n_in; (void)out_size; (void)ws_size;

  const float* x    = (const float*)d_in[0];
  const int*   eidx = (const int*)  d_in[1];
  const float* W0   = (const float*)d_in[2];
  const float* as0  = (const float*)d_in[3];
  const float* ad0  = (const float*)d_in[4];
  const float* b0   = (const float*)d_in[5];
  const float* W1   = (const float*)d_in[6];
  const float* as1  = (const float*)d_in[7];
  const float* ad1  = (const float*)d_in[8];
  const float* b1   = (const float*)d_in[9];
  const float* W2   = (const float*)d_in[10];
  const float* as2  = (const float*)d_in[11];
  const float* ad2  = (const float*)d_in[12];
  const float* b2   = (const float*)d_in[13];
  const float* ln0g = (const float*)d_in[14];
  const float* ln0b = (const float*)d_in[15];
  const float* ln1g = (const float*)d_in[16];
  const float* ln1b = (const float*)d_in[17];
  const float* r0W  = (const float*)d_in[18];
  const float* r0b  = (const float*)d_in[19];
  const float* r1W  = (const float*)d_in[20];
  const float* r1b  = (const float*)d_in[21];
  float* out = (float*)d_out;

  const int* esrc = eidx;
  const int* edst = eidx + N_EDGES;
  const int Etot  = N_EDGES + N_NODES;   // with self-loops

  // ---- workspace carve-out (256B-aligned bump allocator) ----
  char* wp = (char*)d_ws;
  auto alloc = [&](size_t bytes) -> void* {
    void* p = (void*)wp; wp += (bytes + 255) & ~(size_t)255; return p;
  };
  __bf16* x_bf   = (__bf16*)alloc((size_t)N_NODES * IN_DIM * 2);
  __bf16* W0_bf  = (__bf16*)alloc((size_t)IN_DIM * 4 * HID * 2);
  __bf16* W1_bf  = (__bf16*)alloc((size_t)HID * 4 * HID * 2);
  __bf16* W2_bf  = (__bf16*)alloc((size_t)HID * OUT_DIM * 2);
  __bf16* r0W_bf = (__bf16*)alloc((size_t)IN_DIM * HID * 2);
  __bf16* r1W_bf = (__bf16*)alloc((size_t)HID * HID * 2);
  float*  xh     = (float*) alloc((size_t)N_NODES * 4 * HID * 4);
  float*  al_s   = (float*) alloc((size_t)N_NODES * 4 * 4);
  float*  al_d   = (float*) alloc((size_t)N_NODES * 4 * 4);
  float*  denom  = (float*) alloc((size_t)N_NODES * 4 * 4);
  unsigned* mkey = (unsigned*)alloc((size_t)N_NODES * 4 * 4);
  float*  exbuf  = (float*) alloc((size_t)Etot * 4 * 4);
  float*  agg    = (float*) alloc((size_t)N_NODES * 4 * HID * 4);
  float*  resid  = (float*) alloc((size_t)N_NODES * HID * 4);
  __bf16* h_bf   = (__bf16*)alloc((size_t)N_NODES * HID * 2);

  auto cdiv = [](long long a, long long b) -> int { return (int)((a + b - 1) / b); };
  const int TB = 256;

  // ---- fp32 -> bf16 conversions ----
  k_cvt_bf16<<<cdiv((long long)N_NODES * IN_DIM, TB), TB, 0, stream>>>(x, x_bf, N_NODES * IN_DIM);
  k_cvt_bf16<<<cdiv(IN_DIM * 4 * HID, TB), TB, 0, stream>>>(W0, W0_bf, IN_DIM * 4 * HID);
  k_cvt_bf16<<<cdiv(HID * 4 * HID, TB), TB, 0, stream>>>(W1, W1_bf, HID * 4 * HID);
  k_cvt_bf16<<<cdiv(HID * OUT_DIM, TB), TB, 0, stream>>>(W2, W2_bf, HID * OUT_DIM);
  k_cvt_bf16<<<cdiv(IN_DIM * HID, TB), TB, 0, stream>>>(r0W, r0W_bf, IN_DIM * HID);
  k_cvt_bf16<<<cdiv(HID * HID, TB), TB, 0, stream>>>(r1W, r1W_bf, HID * HID);

  const int Mtiles = N_NODES / 16;   // 1250

  // ================= layer 0 (256 -> 4x128, heads=4) =================
  {
    const int H = 4, C = HID, Ng = 4 * HID;
    k_gemm_bf16<<<dim3(Mtiles, Ng / 64), 128, 0, stream>>>(x_bf, W0_bf, nullptr, xh, N_NODES, Ng, IN_DIM);
    k_attn_logits<<<cdiv(N_NODES * H, TB), TB, 0, stream>>>(xh, as0, ad0, al_s, al_d, N_NODES, H, C);
    k_fill_u32<<<cdiv(N_NODES * H, TB), TB, 0, stream>>>(mkey, 0u, N_NODES * H);
    k_fill_f32<<<cdiv(N_NODES * H, TB), TB, 0, stream>>>(denom, 0.f, N_NODES * H);
    k_fill_f32<<<cdiv((long long)N_NODES * H * C, TB), TB, 0, stream>>>(agg, 0.f, N_NODES * H * C);
    k_edge_max<<<cdiv((long long)Etot * H, TB), TB, 0, stream>>>(esrc, edst, N_EDGES, Etot, al_s, al_d, mkey, H);
    k_edge_expsum<<<cdiv((long long)Etot * H, TB), TB, 0, stream>>>(esrc, edst, N_EDGES, Etot, al_s, al_d, mkey, exbuf, denom, H);
    k_edge_agg<<<cdiv((long long)Etot * H * C, TB), TB, 0, stream>>>(esrc, edst, N_EDGES, Etot, exbuf, denom, xh, agg, H, C);
    k_gemm_bf16<<<dim3(Mtiles, HID / 64), 128, 0, stream>>>(x_bf, r0W_bf, r0b, resid, N_NODES, HID, IN_DIM);
    k_post<<<N_NODES, 128, 0, stream>>>(agg, b0, ln0g, ln0b, resid, h_bf, H, C);
  }

  // ================= layer 1 (128 -> 4x128, heads=4) =================
  {
    const int H = 4, C = HID, Ng = 4 * HID;
    k_gemm_bf16<<<dim3(Mtiles, Ng / 64), 128, 0, stream>>>(h_bf, W1_bf, nullptr, xh, N_NODES, Ng, HID);
    k_attn_logits<<<cdiv(N_NODES * H, TB), TB, 0, stream>>>(xh, as1, ad1, al_s, al_d, N_NODES, H, C);
    k_fill_u32<<<cdiv(N_NODES * H, TB), TB, 0, stream>>>(mkey, 0u, N_NODES * H);
    k_fill_f32<<<cdiv(N_NODES * H, TB), TB, 0, stream>>>(denom, 0.f, N_NODES * H);
    k_fill_f32<<<cdiv((long long)N_NODES * H * C, TB), TB, 0, stream>>>(agg, 0.f, N_NODES * H * C);
    k_edge_max<<<cdiv((long long)Etot * H, TB), TB, 0, stream>>>(esrc, edst, N_EDGES, Etot, al_s, al_d, mkey, H);
    k_edge_expsum<<<cdiv((long long)Etot * H, TB), TB, 0, stream>>>(esrc, edst, N_EDGES, Etot, al_s, al_d, mkey, exbuf, denom, H);
    k_edge_agg<<<cdiv((long long)Etot * H * C, TB), TB, 0, stream>>>(esrc, edst, N_EDGES, Etot, exbuf, denom, xh, agg, H, C);
    k_gemm_bf16<<<dim3(Mtiles, HID / 64), 128, 0, stream>>>(h_bf, r1W_bf, r1b, resid, N_NODES, HID, HID);
    // k_post reads only agg/resid and overwrites h_bf -> safe reuse after the GEMMs above
    k_post<<<N_NODES, 128, 0, stream>>>(agg, b1, ln1g, ln1b, resid, h_bf, H, C);
  }

  // ================= layer 2 (128 -> 64, heads=1) =================
  {
    const int H = 1, C = OUT_DIM;
    k_gemm_bf16<<<dim3(Mtiles, 1), 128, 0, stream>>>(h_bf, W2_bf, nullptr, xh, N_NODES, OUT_DIM, HID);
    k_attn_logits<<<cdiv(N_NODES * H, TB), TB, 0, stream>>>(xh, as2, ad2, al_s, al_d, N_NODES, H, C);
    k_fill_u32<<<cdiv(N_NODES * H, TB), TB, 0, stream>>>(mkey, 0u, N_NODES * H);
    k_fill_f32<<<cdiv(N_NODES * H, TB), TB, 0, stream>>>(denom, 0.f, N_NODES * H);
    k_fill_f32<<<cdiv((long long)N_NODES * H * C, TB), TB, 0, stream>>>(agg, 0.f, N_NODES * H * C);
    k_edge_max<<<cdiv((long long)Etot * H, TB), TB, 0, stream>>>(esrc, edst, N_EDGES, Etot, al_s, al_d, mkey, H);
    k_edge_expsum<<<cdiv((long long)Etot * H, TB), TB, 0, stream>>>(esrc, edst, N_EDGES, Etot, al_s, al_d, mkey, exbuf, denom, H);
    k_edge_agg<<<cdiv((long long)Etot * H * C, TB), TB, 0, stream>>>(esrc, edst, N_EDGES, Etot, exbuf, denom, xh, agg, H, C);
    k_final<<<cdiv((long long)N_NODES * C, TB), TB, 0, stream>>>(agg, b2, out, N_NODES, C);
  }
}
